// TaggingItems_34402688041651
// MI455X (gfx1250) — matrix operand order repaired and checked by reading, not compiled, not run on hardware
//
#include <hip/hip_runtime.h>

#define HD 128
#define NLAYERS 4
#define NITEM 10000
#define NTAG 2000
#define VOCAB 50000
#define KTAG 32
#define KITEM 8

typedef __attribute__((ext_vector_type(16))) __bf16 bf16x16;
typedef __attribute__((ext_vector_type(8)))  __bf16 bf16x8;
typedef __attribute__((ext_vector_type(8)))  float  f32x8;

union ABu { bf16x16 v; bf16x8 h[2]; };

// A operand: 16x32 bf16 tile (MxK) from row-major [16][128] buffer (LDS).
// lane: m = lane&15 (row M), g = lane>>4.
// VGPR halves 0..7 : K = kb*32 + 8g + [0..8)
// VGPR halves 8..15: K = kb*32 + 16 + 8g + [0..8)
static __device__ __forceinline__ bf16x16 ld_a(const __bf16* base, int m, int g, int kb) {
  ABu u;
  u.h[0] = *(const bf16x8*)(base + m * 128 + kb * 32 + g * 8);
  u.h[1] = *(const bf16x8*)(base + m * 128 + kb * 32 + 16 + g * 8);
  return u.v;
}

// B operand: 32x16 bf16 tile (KxN) where B = W^T, W row-major [cols][128].
// lane: n = lane&15 -> output column j; g = lane>>4 selects K half.
// lane holds W[j][kb*32 + 8g .. +8) and W[j][kb*32 + 16 + 8g .. +8).
static __device__ __forceinline__ bf16x16 ld_b(const __bf16* __restrict__ W, int j, int g, int kb) {
  ABu u;
  u.h[0] = *(const bf16x8*)(W + j * 128 + kb * 32 + g * 8);
  u.h[1] = *(const bf16x8*)(W + j * 128 + kb * 32 + 16 + g * 8);
  return u.v;
}

static __device__ __forceinline__ f32x8 wmma_bf16(bf16x16 a, bf16x16 b, f32x8 c) {
  return __builtin_amdgcn_wmma_f32_16x16x32_bf16(false, a, false, b, (short)0, c, false, false);
}

static __device__ __forceinline__ float sigm(float x) { return 1.0f / (1.0f + __expf(-x)); }

// ---------------------------------------------------------------------------
// Per-etype argument pack (so one launch can run both etypes concurrently).
// ---------------------------------------------------------------------------
struct SageArgs {
  const __bf16* src;     // [n_src][128] neighbor features
  const __bf16* dst;     // [n_dst][128] self features
  const int*    nbr;     // [n_dst][K]
  const int*    lens;    // [n_dst]
  int           K;
  int           ntiles;  // n_dst / 16
  const __bf16* Wih;     // [512][128] bf16
  const __bf16* Whh;     // [512][128] bf16
  const float*  bih;     // [512]
  const float*  bhh;     // [512]
  const __bf16* Wself;   // [128][128] bf16
  const __bf16* Wneigh;  // [128][128] bf16
  const float*  bias;    // [128]
  const float*  gamma;   // [128] or null
  const float*  beta;    // [128] or null
  __bf16*       outp;    // [n_dst][128]
};

// ---------------------------------------------------------------------------
// Fused LSTM aggregation + SAGE epilogue (+ optional LayerNorm).
// One workgroup (8 waves) owns 16 destination nodes; blocks [0, pa.ntiles)
// run etype A (tags, K=32), the rest run etype B (items, K=8).
// Gate weights: each wave pins its 4 column tiles' B operands (32 x bf16x16 =
// 256 VGPRs) in registers for the whole recurrence — no weight reloads on the
// serial critical path. Neighbor gather is double-buffered/software-pipelined.
// ---------------------------------------------------------------------------
__global__ __launch_bounds__(256)
void sage_dual_kernel(SageArgs pa, SageArgs pb) {
  __shared__ __attribute__((aligned(32))) __bf16 sX0[16 * 128]; // x ping
  __shared__ __attribute__((aligned(32))) __bf16 sX1[16 * 128]; // x pong
  __shared__ __attribute__((aligned(32))) __bf16 sH[16 * 128];  // h state (bf16)
  __shared__ float sC[16 * 128];                                // c state (f32)
  __shared__ float sG[16 * 512];                                // gate pre-activations
  __shared__ float sB[512];                                     // bih + bhh
  __shared__ int   sLen[16];
  __shared__ float sMu[16], sRs[16];

  const bool firstSide = ((int)blockIdx.x < pa.ntiles);
  const SageArgs& p = firstSide ? pa : pb;
  const int tile = firstSide ? (int)blockIdx.x : (int)blockIdx.x - pa.ntiles;

  const int tid  = threadIdx.x;
  const int lane = tid & 31;
  const int wave = tid >> 5;
  const int m    = lane & 15;
  const int g    = lane >> 4;
  const int node0 = tile * 16;
  const int row16 = tid >> 4, seg = tid & 15;   // gather: 16B per thread
  const int K = p.K;

  for (int i = tid; i < 16 * 128; i += 256) { sH[i] = (__bf16)0.0f; sC[i] = 0.0f; }
  for (int i = tid; i < 512; i += 256) sB[i] = p.bih[i] + p.bhh[i];
  if (tid < 16) sLen[tid] = p.lens[node0 + tid];

  // Pin this wave's gate-weight tiles in registers for the whole recurrence:
  // wI/wH[kb][q] is the 32x16 B tile for k-block kb, column tile wave*4+q.
  bf16x16 wI[4][4], wH[4][4];
  #pragma unroll
  for (int kb = 0; kb < 4; ++kb) {
    #pragma unroll
    for (int q = 0; q < 4; ++q) {
      int j = (wave * 4 + q) * 16 + m;
      wI[kb][q] = ld_b(p.Wih, j, g, kb);
      wH[kb][q] = ld_b(p.Whh, j, g, kb);
    }
  }

  // prime pipeline: gather x_0
  {
    int s = p.nbr[(node0 + row16) * K];
    ((uint4*)sX0)[row16 * 16 + seg] = ((const uint4*)(p.src + (size_t)s * 128))[seg];
  }
  __syncthreads();

  for (int t = 0; t < K; ++t) {
    const __bf16* xc = (t & 1) ? sX1 : sX0;   // runtime select, no ptr array
    __bf16*       xn = (t & 1) ? sX0 : sX1;

    // issue neighbor loads for step t+1 now; ds_store after the WMMAs
    uint4 pref = {};
    const bool havePref = (t + 1 < K);
    if (havePref) {
      int s = p.nbr[(node0 + row16) * K + (t + 1)];
      pref = ((const uint4*)(p.src + (size_t)s * 128))[seg];
    }

    // gate GEMM: g[16,512] = x @ Wih^T + h @ Whh^T ; wave owns 4 column tiles
    f32x8 acc[4] = {};
    #pragma unroll
    for (int kb = 0; kb < 4; ++kb) {
      bf16x16 aX = ld_a(xc, m, g, kb);
      bf16x16 aH = ld_a(sH, m, g, kb);
      #pragma unroll
      for (int q = 0; q < 4; ++q) {
        acc[q] = wmma_bf16(aX, wI[kb][q], acc[q]);
        acc[q] = wmma_bf16(aH, wH[kb][q], acc[q]);
      }
    }
    #pragma unroll
    for (int q = 0; q < 4; ++q) {
      int col = (wave * 4 + q) * 16 + m;
      #pragma unroll
      for (int r = 0; r < 8; ++r) sG[(r + 8 * g) * 512 + col] = acc[q][r];
    }
    if (havePref) ((uint4*)xn)[row16 * 16 + seg] = pref;
    __syncthreads();

    // gate nonlinearity + state update (PyTorch order i,f,g,o), mask pads
    for (int e = tid; e < 16 * 128; e += 256) {
      int row = e >> 7, col = e & 127;
      if (t < sLen[row]) {
        float gi = sG[row * 512 + col]       + sB[col];
        float gf = sG[row * 512 + col + 128] + sB[col + 128];
        float gg = sG[row * 512 + col + 256] + sB[col + 256];
        float go = sG[row * 512 + col + 384] + sB[col + 384];
        float iv = sigm(gi), fv = sigm(gf), ov = sigm(go);
        float gv = tanhf(gg);
        float cn = fv * sC[e] + iv * gv;
        sC[e] = cn;
        sH[e] = (__bf16)(ov * tanhf(cn));
      }
    }
    __syncthreads();
  }

  // SAGE epilogue: new = h_dst @ Wself^T + hn @ Wneigh^T + b  (hn == sH)
  ((uint4*)sX0)[row16 * 16 + seg] =
      ((const uint4*)(p.dst + (size_t)(node0 + row16) * 128))[seg];
  __syncthreads();
  {
    f32x8 acc = {};
    const int j = wave * 16 + m;
    #pragma unroll
    for (int kb = 0; kb < 4; ++kb) {
      acc = wmma_bf16(ld_a(sX0, m, g, kb), ld_b(p.Wself,  j, g, kb), acc);
      acc = wmma_bf16(ld_a(sH,  m, g, kb), ld_b(p.Wneigh, j, g, kb), acc);
    }
    float bb = p.bias[j];
    #pragma unroll
    for (int r = 0; r < 8; ++r) sG[(r + 8 * g) * 128 + j] = acc[r] + bb;
  }
  __syncthreads();

  if (p.gamma != nullptr) {  // fused LayerNorm over H
    if (tid < 16) {
      float s = 0.f;
      for (int c = 0; c < 128; ++c) s += sG[tid * 128 + c];
      float mu = s * (1.0f / 128.0f);
      float v = 0.f;
      for (int c = 0; c < 128; ++c) { float d = sG[tid * 128 + c] - mu; v += d * d; }
      sMu[tid] = mu;
      sRs[tid] = rsqrtf(v * (1.0f / 128.0f) + 1e-5f);
    }
    __syncthreads();
    for (int e = tid; e < 16 * 128; e += 256) {
      int row = e >> 7, col = e & 127;
      float val = (sG[row * 128 + col] - sMu[row]) * sRs[row] * p.gamma[col] + p.beta[col];
      p.outp[(size_t)(node0 + row) * 128 + col] = (__bf16)val;
    }
  } else {
    for (int e = tid; e < 16 * 128; e += 256) {
      int row = e >> 7, col = e & 127;
      p.outp[(size_t)(node0 + row) * 128 + col] = (__bf16)sG[row * 128 + col];
    }
  }
}

// ---------------------------------------------------------------------------
// Final: tmp = leaky_relu(h_item @ W_final^T)
// ---------------------------------------------------------------------------
__global__ __launch_bounds__(256)
void final_gemm_kernel(const __bf16* __restrict__ hitem,
                       const __bf16* __restrict__ Wf,   // [128][128] bf16
                       float* __restrict__ tmp)         // [NITEM][128]
{
  __shared__ __attribute__((aligned(32))) __bf16 sX[16 * 128];
  const int tid = threadIdx.x, lane = tid & 31, wave = tid >> 5;
  const int m = lane & 15, g = lane >> 4;
  const int node0 = blockIdx.x * 16;
  {
    int row = tid >> 4, seg = tid & 15;
    ((uint4*)sX)[row * 16 + seg] =
        ((const uint4*)(hitem + (size_t)(node0 + row) * 128))[seg];
  }
  __syncthreads();
  f32x8 acc = {};
  const int j = wave * 16 + m;
  #pragma unroll
  for (int kb = 0; kb < 4; ++kb)
    acc = wmma_bf16(ld_a(sX, m, g, kb), ld_b(Wf, j, g, kb), acc);
  #pragma unroll
  for (int r = 0; r < 8; ++r) {
    float v = acc[r];
    v = v > 0.f ? v : 0.01f * v;
    tmp[(size_t)(node0 + r + 8 * g) * 128 + j] = v;
  }
}

// ---------------------------------------------------------------------------
// unique() machinery + small utilities
// ---------------------------------------------------------------------------
__global__ void k_f32_to_bf16(const float* __restrict__ in, __bf16* __restrict__ out, int n) {
  int i = blockIdx.x * blockDim.x + threadIdx.x;
  if (i < n) out[i] = (__bf16)in[i];
}

__global__ void k_zero_i32(int* __restrict__ p, int n) {
  int i = blockIdx.x * blockDim.x + threadIdx.x;
  if (i < n) p[i] = 0;
}

__global__ void k_mark(const int* __restrict__ items, int* __restrict__ present, int n) {
  int i = blockIdx.x * blockDim.x + threadIdx.x;
  if (i < n) present[items[i]] = 1;
}

// single-block exclusive scan over `n` ints (Hillis-Steele per 1024 chunk)
__global__ __launch_bounds__(1024)
void k_scan(const int* __restrict__ present, int* __restrict__ rankv, int n) {
  __shared__ int buf[1024];
  const int tid = threadIdx.x;
  int offset = 0;
  for (int base = 0; base < n; base += 1024) {
    int idx = base + tid;
    int v = (idx < n) ? present[idx] : 0;
    buf[tid] = v;
    __syncthreads();
    for (int s = 1; s < 1024; s <<= 1) {
      int t = (tid >= s) ? buf[tid - s] : 0;
      __syncthreads();
      buf[tid] += t;
      __syncthreads();
    }
    if (idx < n) rankv[idx] = offset + buf[tid] - v;   // exclusive
    int tot = buf[1023];
    __syncthreads();
    offset += tot;
  }
}

__global__ void k_scatter_uq(const int* __restrict__ present, const int* __restrict__ rankv,
                             int* __restrict__ uq, int vocab) {
  int v = blockIdx.x * blockDim.x + threadIdx.x;
  if (v < vocab && present[v]) {
    int r = rankv[v];
    if (r < NITEM) uq[r] = v;
  }
}

__global__ void k_inv(const int* __restrict__ items, const int* __restrict__ rankv,
                      int* __restrict__ inv, int n) {
  int i = blockIdx.x * blockDim.x + threadIdx.x;
  if (i < n) inv[i] = rankv[items[i]];
}

__global__ void k_gather_rows_bf16(const float* __restrict__ table, const int* __restrict__ idx,
                                   __bf16* __restrict__ out, int nelem) {
  int i = blockIdx.x * blockDim.x + threadIdx.x;
  if (i < nelem) {
    int r = i >> 7, c = i & 127;
    out[i] = (__bf16)table[(size_t)idx[r] * 128 + c];
  }
}

__global__ void k_gather_out(const float* __restrict__ tmp, const int* __restrict__ inv,
                             float* __restrict__ out, int nelem) {
  int i = blockIdx.x * blockDim.x + threadIdx.x;
  if (i < nelem) out[i] = tmp[(size_t)inv[i >> 7] * 128 + (i & 127)];
}

// ---------------------------------------------------------------------------
extern "C" void kernel_launch(void* const* d_in, const int* in_sizes, int n_in,
                              void* d_out, int out_size, void* d_ws, size_t ws_size,
                              hipStream_t stream) {
  const int*   items      = (const int*)d_in[0];
  const int*   nbr_item   = (const int*)d_in[1];
  const int*   len_tag    = (const int*)d_in[2];
  const int*   nbr_tag    = (const int*)d_in[3];
  const int*   len_item   = (const int*)d_in[4];
  const float* item_table = (const float*)d_in[5];
  const float* tag_table  = (const float*)d_in[6];
  const float* Wih_as     = (const float*)d_in[7];
  const float* Whh_as     = (const float*)d_in[8];
  const float* bih_as     = (const float*)d_in[9];
  const float* bhh_as     = (const float*)d_in[10];
  const float* Wself_as   = (const float*)d_in[11];
  const float* Wneigh_as  = (const float*)d_in[12];
  const float* b_as       = (const float*)d_in[13];
  const float* Wih_ras    = (const float*)d_in[14];
  const float* Whh_ras    = (const float*)d_in[15];
  const float* bih_ras    = (const float*)d_in[16];
  const float* bhh_ras    = (const float*)d_in[17];
  const float* Wself_ras  = (const float*)d_in[18];
  const float* Wneigh_ras = (const float*)d_in[19];
  const float* b_ras      = (const float*)d_in[20];
  const float* ln_gamma   = (const float*)d_in[21];
  const float* ln_beta    = (const float*)d_in[22];
  const float* W_final    = (const float*)d_in[23];
  float* out = (float*)d_out;

  // workspace layout (256B aligned)
  size_t off = 0;
  auto A = [&](size_t bytes) -> size_t {
    size_t o = off; off += (bytes + 255) & ~(size_t)255; return o;
  };
  char* ws = (char*)d_ws;
  const size_t nIH = (size_t)NLAYERS * 512 * 128;   // [L,4H,H] elems
  const size_t nSQ = (size_t)NLAYERS * 128 * 128;   // [L,H,H] elems

  __bf16* wih_as_b  = (__bf16*)(ws + A(nIH * 2));
  __bf16* whh_as_b  = (__bf16*)(ws + A(nIH * 2));
  __bf16* wsf_as_b  = (__bf16*)(ws + A(nSQ * 2));
  __bf16* wng_as_b  = (__bf16*)(ws + A(nSQ * 2));
  __bf16* wih_rs_b  = (__bf16*)(ws + A(nIH * 2));
  __bf16* whh_rs_b  = (__bf16*)(ws + A(nIH * 2));
  __bf16* wsf_rs_b  = (__bf16*)(ws + A(nSQ * 2));
  __bf16* wng_rs_b  = (__bf16*)(ws + A(nSQ * 2));
  __bf16* wfin_b    = (__bf16*)(ws + A((size_t)128 * 128 * 2));
  int*    present   = (int*)   (ws + A((size_t)VOCAB * 4));
  int*    rankv     = (int*)   (ws + A((size_t)VOCAB * 4));
  int*    uq        = (int*)   (ws + A((size_t)NITEM * 4));
  int*    inv       = (int*)   (ws + A((size_t)NITEM * 4));
  __bf16* hitem_b0  = (__bf16*)(ws + A((size_t)NITEM * 128 * 2));
  __bf16* hitem_b1  = (__bf16*)(ws + A((size_t)NITEM * 128 * 2));
  __bf16* htag_b0   = (__bf16*)(ws + A((size_t)NTAG * 128 * 2));
  __bf16* htag_b1   = (__bf16*)(ws + A((size_t)NTAG * 128 * 2));
  float*  tmp       = (float*) (ws + A((size_t)NITEM * 128 * 4));
  __bf16* hitem_b[2] = { hitem_b0, hitem_b1 };
  __bf16* htag_b[2]  = { htag_b0,  htag_b1 };

  auto G = [](int n, int b) { return (n + b - 1) / b; };

  // 1) weights -> bf16 (same row-major layouts)
  k_f32_to_bf16<<<G((int)nIH, 256), 256, 0, stream>>>(Wih_as,    wih_as_b, (int)nIH);
  k_f32_to_bf16<<<G((int)nIH, 256), 256, 0, stream>>>(Whh_as,    whh_as_b, (int)nIH);
  k_f32_to_bf16<<<G((int)nSQ, 256), 256, 0, stream>>>(Wself_as,  wsf_as_b, (int)nSQ);
  k_f32_to_bf16<<<G((int)nSQ, 256), 256, 0, stream>>>(Wneigh_as, wng_as_b, (int)nSQ);
  k_f32_to_bf16<<<G((int)nIH, 256), 256, 0, stream>>>(Wih_ras,   wih_rs_b, (int)nIH);
  k_f32_to_bf16<<<G((int)nIH, 256), 256, 0, stream>>>(Whh_ras,   whh_rs_b, (int)nIH);
  k_f32_to_bf16<<<G((int)nSQ, 256), 256, 0, stream>>>(Wself_ras, wsf_rs_b, (int)nSQ);
  k_f32_to_bf16<<<G((int)nSQ, 256), 256, 0, stream>>>(Wneigh_ras,wng_rs_b, (int)nSQ);
  k_f32_to_bf16<<<G(128 * 128, 256), 256, 0, stream>>>(W_final,  wfin_b, 128 * 128);

  // 2) unique(items): present -> exclusive scan -> uq/inv
  k_zero_i32<<<G(VOCAB, 256), 256, 0, stream>>>(present, VOCAB);
  k_zero_i32<<<G(NITEM, 256), 256, 0, stream>>>(uq, NITEM);   // fill_value = 0
  k_mark<<<G(NITEM, 256), 256, 0, stream>>>(items, present, NITEM);
  k_scan<<<1, 1024, 0, stream>>>(present, rankv, VOCAB);
  k_scatter_uq<<<G(VOCAB, 256), 256, 0, stream>>>(present, rankv, uq, VOCAB);
  k_inv<<<G(NITEM, 256), 256, 0, stream>>>(items, rankv, inv, NITEM);

  // 3) initial features (bf16)
  k_gather_rows_bf16<<<G(NITEM * 128, 256), 256, 0, stream>>>(item_table, uq, hitem_b[0], NITEM * 128);
  k_f32_to_bf16<<<G(NTAG * 128, 256), 256, 0, stream>>>(tag_table, htag_b[0], NTAG * 128);

  // 4) layers: 'as' (item->tag, K=32) and 'ras' (tag->item, K=8) fused into
  //    one launch so the short recurrence fills the machine alongside the long one.
  int cur = 0;
  for (int l = 0; l < NLAYERS; ++l) {
    SageArgs pa;
    pa.src = hitem_b[cur]; pa.dst = htag_b[cur];
    pa.nbr = nbr_item; pa.lens = len_tag; pa.K = KTAG; pa.ntiles = NTAG / 16;
    pa.Wih = wih_as_b + (size_t)l * 512 * 128;
    pa.Whh = whh_as_b + (size_t)l * 512 * 128;
    pa.bih = bih_as + (size_t)l * 512;
    pa.bhh = bhh_as + (size_t)l * 512;
    pa.Wself  = wsf_as_b + (size_t)l * 128 * 128;
    pa.Wneigh = wng_as_b + (size_t)l * 128 * 128;
    pa.bias = b_as + (size_t)l * 128;
    pa.gamma = nullptr; pa.beta = nullptr;
    pa.outp = htag_b[1 - cur];

    SageArgs pb;
    pb.src = htag_b[cur]; pb.dst = hitem_b[cur];
    pb.nbr = nbr_tag; pb.lens = len_item; pb.K = KITEM; pb.ntiles = NITEM / 16;
    pb.Wih = wih_rs_b + (size_t)l * 512 * 128;
    pb.Whh = whh_rs_b + (size_t)l * 512 * 128;
    pb.bih = bih_ras + (size_t)l * 512;
    pb.bhh = bhh_ras + (size_t)l * 512;
    pb.Wself  = wsf_rs_b + (size_t)l * 128 * 128;
    pb.Wneigh = wng_rs_b + (size_t)l * 128 * 128;
    pb.bias = b_ras + (size_t)l * 128;
    pb.gamma = ln_gamma + (size_t)l * 128;
    pb.beta  = ln_beta + (size_t)l * 128;
    pb.outp = hitem_b[1 - cur];

    sage_dual_kernel<<<pa.ntiles + pb.ntiles, 256, 0, stream>>>(pa, pb);
    cur = 1 - cur;
  }

  // 5) final linear + leaky_relu, then scatter by inv
  final_gemm_kernel<<<NITEM / 16, 256, 0, stream>>>(hitem_b[cur], wfin_b, tmp);
  k_gather_out<<<G(NITEM * 128, 256), 256, 0, stream>>>(tmp, inv, out, NITEM * 128);
}
